// VSSM_54271206752691
// MI455X (gfx1250) — matrix operand order repaired and checked by reading, not compiled, not run on hardware
//
#include <hip/hip_runtime.h>
#include <hip/hip_bf16.h>
#include <cstddef>
#include <cstdint>

// ---------------------------------------------------------------------------
// VSSM block for MI455X (gfx1250, wave32, WMMA + TDM).
//   DIM=96 EXPAND=2 D_STATE=16 INNER=192 D_INNER=384 DT_RANK=12 K=4
//   B=2 H=56 W=56 L=3136
// Dense projections: v_wmma_f32_16x16x32_f16, B tile staged to LDS by the
// Tensor Data Mover (tensor_load_to_lds + s_wait_tensorcnt), 4 waves/block
// stacked in M share the tile. Selective scan: 16 lanes = 16 states/channel.
// xs (4-direction tensor) is never materialized (index gather).
// ---------------------------------------------------------------------------

typedef __attribute__((ext_vector_type(16))) _Float16 v16h;
typedef __attribute__((ext_vector_type(8)))  float    v8f;
typedef __attribute__((ext_vector_type(4)))  unsigned int u32x4;
typedef __attribute__((ext_vector_type(8)))  int      i32x8;
typedef __attribute__((ext_vector_type(4)))  int      i32x4;

#define Bc   2
#define Hh   56
#define Ww   56
#define Ll   3136      // H*W
#define C0   96
#define INN  192
#define DIN  384
#define K4   4
#define NST  16
#define DTR  12
#define XDR  44        // DT_RANK + 2*D_STATE

__device__ __forceinline__ float silu_f(float x) {
    return x * (1.0f / (1.0f + __expf(-x)));
}
__device__ __forceinline__ float softplus_f(float x) {
    return (x > 20.0f) ? x : log1pf(__expf(x));
}

// direction-local position l -> source index into row-major (h*W+w) flat seq
__device__ __forceinline__ int dir_src(int k, int l) {
    if (k == 0) return l;
    if (k == 2) return (Ll - 1) - l;
    int j = (k == 1) ? l : (Ll - 1) - l;     // k==1 or k==3
    int w_ = j / Hh;                          // j = w*H + h
    int h_ = j - w_ * Hh;
    return h_ * Ww + w_;
}

__device__ __forceinline__ v16h a_frag_from_f32(const float* ap) {
    float4 f0 = *(const float4*)(ap);
    float4 f1 = *(const float4*)(ap + 4);
    float4 f2 = *(const float4*)(ap + 16);
    float4 f3 = *(const float4*)(ap + 20);
    v16h af;
    af[0]=(_Float16)f0.x;  af[1]=(_Float16)f0.y;  af[2]=(_Float16)f0.z;  af[3]=(_Float16)f0.w;
    af[4]=(_Float16)f1.x;  af[5]=(_Float16)f1.y;  af[6]=(_Float16)f1.z;  af[7]=(_Float16)f1.w;
    af[8]=(_Float16)f2.x;  af[9]=(_Float16)f2.y;  af[10]=(_Float16)f2.z; af[11]=(_Float16)f2.w;
    af[12]=(_Float16)f3.x; af[13]=(_Float16)f3.y; af[14]=(_Float16)f3.z; af[15]=(_Float16)f3.w;
    return af;
}

// ---------------------------------------------------------------------------
// WMMA GEMM with TDM-staged B tile: C[M,N] = act(A[M,K] @ B[K,N] + bias)
// Block = 128 threads (4 waves stacked in M, block tile 64x16).
// Per K-slab (32): TDM DMAs the 32x16 f32 B tile into LDS (one issue,
// TENSORcnt-tracked), all 4 waves consume it. M%64==0, N%16==0, K%32==0.
// ---------------------------------------------------------------------------
template <int ACT>  // 0 = none, 1 = silu
__global__ void gemm_wmma(const float* __restrict__ A, const float* __restrict__ Bm,
                          const float* __restrict__ bias, float* __restrict__ Cmat,
                          int M, int N, int Kd) {
    __shared__ float ldsB[32 * 16];            // one K-slab of the B tile (2 KB)

    const int lane  = threadIdx.x & 31;
    const int wave  = threadIdx.x >> 5;        // 0..3
    const int half  = lane >> 4;               // 0|1
    const int rl    = lane & 15;
    const int tileN = blockIdx.x * 16;
    const int tileM = blockIdx.y * 64 + wave * 16;

    const unsigned lds_off = (unsigned)(size_t)(&ldsB[0]);

    // D# group1 (static): wg_mask=0, data_size=2 (4B), tensor_dim0=N,
    // tensor_dim1=Kd, tile_dim0=16, tile_dim1=32, dim0_stride=N.
    i32x8 g1;
    g1[0] = (2 << 16);
    g1[1] = (N & 0xFFFF) << 16;
    g1[2] = ((unsigned)N >> 16) | ((Kd & 0xFFFF) << 16);
    g1[3] = ((unsigned)Kd >> 16) | (16 << 16);
    g1[4] = 32;                                // tile_dim1=32, tile_dim2=0
    g1[5] = N;                                 // dim0_stride[31:0]
    g1[6] = 0;
    g1[7] = 0;
    const i32x4 z4 = {0, 0, 0, 0};
    const i32x8 z8 = {0, 0, 0, 0, 0, 0, 0, 0};

    v8f acc = {};
    const float* arow = A + (size_t)(tileM + rl) * Kd;

    for (int kb = 0; kb < Kd; kb += 32) {
        // ---- TDM: B[kb:kb+32, tileN:tileN+16] -> LDS (wave 0 issues)
        if (threadIdx.x < 32) {
            uint64_t ga = (uint64_t)(const void*)(Bm + (size_t)kb * N + tileN);
            u32x4 g0;
            g0[0] = 1u;                                            // count=1
            g0[1] = lds_off;                                       // lds_addr
            g0[2] = (unsigned)(ga & 0xFFFFFFFFu);                  // gaddr lo
            g0[3] = (unsigned)((ga >> 32) & 0x01FFFFFFu) | (2u << 30); // hi|type
            __builtin_amdgcn_tensor_load_to_lds(g0, g1, z4, z4, z8, 0);
        }
        __builtin_amdgcn_s_wait_tensorcnt(0);
        __syncthreads();

        // ---- A fragment: 16x32 f16 from global (per-wave rows)
        const float* ap = arow + kb + half * 8;
        v16h af = a_frag_from_f32(ap);
        if (kb + 32 < Kd) __builtin_prefetch(ap + 32, 0, 0);

        // ---- B fragment: 32x16 f16 from LDS (shared across 4 waves)
        v16h bf;
        #pragma unroll
        for (int i = 0; i < 16; ++i)
            bf[i] = (_Float16)ldsB[((i < 8 ? i : i + 8) + half * 8) * 16 + rl];

        acc = __builtin_amdgcn_wmma_f32_16x16x32_f16(false, af, false, bf,
                                                     (short)0, acc, false, false);
        __syncthreads();                       // protect ldsB before next slab
    }

    #pragma unroll
    for (int v = 0; v < 8; ++v) {
        int m  = tileM + v + 8 * half;
        int nn = tileN + rl;
        float val = acc[v];
        if (bias) val += bias[nn];
        if (ACT == 1) val = silu_f(val);
        Cmat[(size_t)m * N + nn] = val;
    }
}

// ---------------------------------------------------------------------------
// xproj GEMM with direction gather (B side permuted -> per-lane loads):
//   xd[b,k,d,l] = sum_c xproj_w[k,d,c] * xi[b,c, g_k(l)]
// M=44 (3 tiles of 16, guarded), K=384, N=L tiled by 16.
// grid = (L/16, 3, B*K4), block = 32
// ---------------------------------------------------------------------------
__global__ void xproj_gemm_wmma(const float* __restrict__ xi,   // (B, DIN, L)
                                const float* __restrict__ xw,   // (K4, 44, DIN)
                                float* __restrict__ xd) {       // (B, K4, 44, L)
    const int lane  = threadIdx.x & 31;
    const int half  = lane >> 4;
    const int rl    = lane & 15;
    const int tileL = blockIdx.x * 16;
    const int tileM = blockIdx.y * 16;
    const int k     = blockIdx.z & 3;
    const int b     = blockIdx.z >> 2;

    const int j   = tileL + rl;
    const int src = dir_src(k, j);

    const float* xib = xi + (size_t)b * DIN * Ll;
    const float* aw  = xw + (size_t)k * XDR * DIN;
    const int row    = tileM + rl;
    const bool rok   = (row < XDR);

    v8f acc = {};
    for (int kb = 0; kb < DIN; kb += 32) {
        v16h af = {};
        if (rok) af = a_frag_from_f32(aw + (size_t)row * DIN + kb + half * 8);
        v16h bf;
        #pragma unroll
        for (int i = 0; i < 16; ++i) {
            int cc = kb + (i < 8 ? i : i + 8) + half * 8;
            bf[i] = (_Float16)xib[(size_t)cc * Ll + src];
        }
        acc = __builtin_amdgcn_wmma_f32_16x16x32_f16(false, af, false, bf,
                                                     (short)0, acc, false, false);
    }

    float* xdb = xd + (size_t)(b * K4 + k) * XDR * Ll;
    #pragma unroll
    for (int v = 0; v < 8; ++v) {
        int m = tileM + v + 8 * half;
        if (m < XDR) xdb[(size_t)m * Ll + tileL + rl] = acc[v];
    }
}

// ---------------------------------------------------------------------------
// Depthwise 3x3 conv (SAME) + bias + SiLU. Input NHWC (row stride inStride),
// output NHWC (stride C) or NCHW flat (B,C,L).
// ---------------------------------------------------------------------------
__global__ void dwconv_silu(const float* __restrict__ in, const float* __restrict__ wgt,
                            const float* __restrict__ bias, float* __restrict__ out,
                            int C, int inStride, int outCHW) {
    int idx = blockIdx.x * blockDim.x + threadIdx.x;
    int total = Bc * Hh * Ww * C;
    if (idx >= total) return;
    int c   = idx % C;
    int rem = idx / C;
    int w_  = rem % Ww; rem /= Ww;
    int h_  = rem % Hh;
    int b   = rem / Hh;

    float acc = bias[c];
    #pragma unroll
    for (int dy = -1; dy <= 1; ++dy) {
        int yy = h_ + dy;
        if (yy < 0 || yy >= Hh) continue;
        #pragma unroll
        for (int dx = -1; dx <= 1; ++dx) {
            int xx = w_ + dx;
            if (xx < 0 || xx >= Ww) continue;
            acc += in[((size_t)(b * Hh + yy) * Ww + xx) * inStride + c]
                 * wgt[c * 9 + (dy + 1) * 3 + (dx + 1)];
        }
    }
    acc = silu_f(acc);
    if (outCHW)
        out[((size_t)b * C + c) * Ll + h_ * Ww + w_] = acc;
    else
        out[((size_t)(b * Hh + h_) * Ww + w_) * C + c] = acc;
}

// ---------------------------------------------------------------------------
// Selective scan. 16 lanes = 16 states of one (b,k,c) channel; 2 channels
// per wave32; 16 channels per 256-thread block. dt computed inline
// (softplus(xd[0:12] . dtp_w + dtp_b)); y reduced with shfl_xor.
// ---------------------------------------------------------------------------
__global__ void scan_kernel(const float* __restrict__ xi,     // (B, DIN, L)
                            const float* __restrict__ xd,     // (B, K4, 44, L)
                            const float* __restrict__ dtp_w,  // (K4, DIN, 12)
                            const float* __restrict__ dtp_b,  // (K4, DIN)
                            const float* __restrict__ A_logs, // (K4, DIN, 16)
                            const float* __restrict__ Ds,     // (K4, DIN)
                            float* __restrict__ oy) {         // (B, K4, DIN, L)
    const int grp = threadIdx.x >> 4;          // 16 groups / block
    const int n   = threadIdx.x & 15;
    const int ch  = blockIdx.x * 16 + grp;     // 0 .. B*K4*DIN-1
    const int c   = ch % DIN;
    const int k   = (ch / DIN) & 3;
    const int b   = ch / (DIN * K4);

    const int kc   = k * DIN + c;
    const float Av = -__expf(A_logs[(kc << 4) + n]);
    float dw[DTR];
    #pragma unroll
    for (int r = 0; r < DTR; ++r) dw[r] = dtp_w[kc * DTR + r];
    const float db = dtp_b[kc];
    const float Dv = Ds[kc];

    const float* xdp = xd + (size_t)(b * K4 + k) * XDR * Ll;
    const float* xip = xi + ((size_t)b * DIN + c) * Ll;
    float*       oyp = oy + ((size_t)(b * K4 + k) * DIN + c) * Ll;

    float h = 0.0f;
    for (int l = 0; l < Ll; ++l) {
        float s = db;
        #pragma unroll
        for (int r = 0; r < DTR; ++r) s += xdp[r * Ll + l] * dw[r];
        float dt = softplus_f(s);

        float u  = xip[dir_src(k, l)];
        float Bn = xdp[(DTR + n) * Ll + l];
        float Cn = xdp[(DTR + NST + n) * Ll + l];

        h = h * __expf(dt * Av) + (dt * u) * Bn;

        float p = h * Cn;
        p += __shfl_xor(p, 8, 32);
        p += __shfl_xor(p, 4, 32);
        p += __shfl_xor(p, 2, 32);
        p += __shfl_xor(p, 1, 32);
        if (n == 0) oyp[l] = p + u * Dv;
    }
}

// ---------------------------------------------------------------------------
// Combine 4 scan directions -> (B, L, DIN) row-major (ready for layernorm).
// ---------------------------------------------------------------------------
__global__ void combine_kernel(const float* __restrict__ oy, float* __restrict__ yc) {
    int idx = blockIdx.x * blockDim.x + threadIdx.x;
    int total = Bc * Ll * DIN;
    if (idx >= total) return;
    int c   = idx % DIN;
    int rem = idx / DIN;
    int l   = rem % Ll;
    int b   = rem / Ll;
    int h_  = l / Ww;
    int w_  = l - h_ * Ww;
    int lt  = w_ * Hh + h_;

    const size_t s0 = ((size_t)(b * K4 + 0) * DIN + c) * Ll;
    const size_t s1 = ((size_t)(b * K4 + 1) * DIN + c) * Ll;
    const size_t s2 = ((size_t)(b * K4 + 2) * DIN + c) * Ll;
    const size_t s3 = ((size_t)(b * K4 + 3) * DIN + c) * Ll;

    float v = oy[s0 + l] + oy[s2 + (Ll - 1 - l)]
            + oy[s1 + lt] + oy[s3 + (Ll - 1 - lt)];
    yc[((size_t)b * Ll + l) * DIN + c] = v;
}

// ---------------------------------------------------------------------------
// In-place LayerNorm: one wave32 per row, shfl_xor reductions.
// ---------------------------------------------------------------------------
__global__ void layernorm_inplace(float* __restrict__ x, const float* __restrict__ g,
                                  const float* __restrict__ bt, int C, int rows) {
    const int wave = threadIdx.x >> 5;
    const int lane = threadIdx.x & 31;
    const int row  = blockIdx.x * (blockDim.x >> 5) + wave;
    if (row >= rows) return;
    float* p = x + (size_t)row * C;

    float s = 0.0f, s2 = 0.0f;
    for (int c = lane; c < C; c += 32) {
        float v = p[c];
        s += v; s2 += v * v;
    }
    #pragma unroll
    for (int o = 16; o > 0; o >>= 1) {
        s  += __shfl_xor(s,  o, 32);
        s2 += __shfl_xor(s2, o, 32);
    }
    float m   = s / C;
    float var = s2 / C - m * m;
    float inv = rsqrtf(var + 1e-5f);
    for (int c = lane; c < C; c += 32) {
        float v = p[c];
        p[c] = (v - m) * inv * g[c] + bt[c];
    }
}

// y_ln * silu(z)  where z lives in the second half of xz rows (stride 768)
__global__ void ymul_kernel(const float* __restrict__ yln, const float* __restrict__ xz,
                            float* __restrict__ o) {
    int idx = blockIdx.x * blockDim.x + threadIdx.x;
    int total = Bc * Ll * DIN;
    if (idx >= total) return;
    int c   = idx % DIN;
    int row = idx / DIN;
    float zv = xz[(size_t)row * (2 * DIN) + DIN + c];
    o[idx] = yln[idx] * silu_f(zv);
}

__global__ void emul_kernel(const float* __restrict__ a, const float* __restrict__ b,
                            float* __restrict__ o, int n) {
    int idx = blockIdx.x * blockDim.x + threadIdx.x;
    if (idx < n) o[idx] = a[idx] * b[idx];
}

// ---------------------------------------------------------------------------
extern "C" void kernel_launch(void* const* d_in, const int* in_sizes, int n_in,
                              void* d_out, int out_size, void* d_ws, size_t ws_size,
                              hipStream_t stream) {
    const float* x        = (const float*)d_in[0];
    const float* w1       = (const float*)d_in[1];
    const float* b1       = (const float*)d_in[2];
    const float* dw_w     = (const float*)d_in[3];
    const float* dw_b     = (const float*)d_in[4];
    const float* inproj_w = (const float*)d_in[5];
    const float* cv_w     = (const float*)d_in[6];
    const float* cv_b     = (const float*)d_in[7];
    const float* xproj_w  = (const float*)d_in[8];
    const float* dtp_w    = (const float*)d_in[9];
    const float* dtp_b    = (const float*)d_in[10];
    const float* A_logs   = (const float*)d_in[11];
    const float* Ds       = (const float*)d_in[12];
    const float* onorm_g  = (const float*)d_in[13];
    const float* onorm_b  = (const float*)d_in[14];
    const float* outp_w   = (const float*)d_in[15];
    const float* norm_g   = (const float*)d_in[16];
    const float* norm_b   = (const float*)d_in[17];
    const float* w2       = (const float*)d_in[18];
    const float* b2       = (const float*)d_in[19];
    const float* wo       = (const float*)d_in[20];
    const float* bo       = (const float*)d_in[21];
    float* out = (float*)d_out;
    float* ws  = (float*)d_ws;

    const int rows = Bc * Ll;                         // 6272 (= 98 * 64)
    // workspace layout (floats); peak ~20.4M floats (~78 MB)
    float* x1pre = ws;                                //  1,204,224
    float* x1    = ws + 1204224;                      //  1,204,224
    float* xz    = ws + 2408448;                      //  4,816,896
    float* xic   = ws + 7225344;                      //  2,408,448 (B,DIN,L)
    float* xd    = ws + 9633792;                      //  1,103,872 (B,K,44,L)
    float* oy    = ws + 10737664;                     //  9,633,792 (B,K,DIN,L)
    float* yc    = ws;                                // reuse x1pre+x1 after scan
    float* ymul  = oy;                                // reuse oy region after combine
    float* tbuf  = oy + 2408448;
    float* x2    = oy + 2408448 + 1204224;
    float* fin   = oy + 2408448 + 2 * 1204224;

    // 1) x @ w1 + b1
    gemm_wmma<0><<<dim3(INN / 16, rows / 64), 128, 0, stream>>>(x, w1, b1, x1pre, rows, INN, C0);
    // 2) depthwise 3x3 + bias + silu (NHWC -> NHWC)
    dwconv_silu<<<(Bc * Ll * INN + 255) / 256, 256, 0, stream>>>(x1pre, dw_w, dw_b, x1, INN, INN, 0);
    // 3) x1 @ inproj_w  -> xz (B*L, 768)
    gemm_wmma<0><<<dim3((2 * DIN) / 16, rows / 64), 128, 0, stream>>>(x1, inproj_w, nullptr, xz, rows, 2 * DIN, INN);
    // 4) depthwise 3x3 + bias + silu on xi half (stride 768) -> (B,DIN,L)
    dwconv_silu<<<(Bc * Ll * DIN + 255) / 256, 256, 0, stream>>>(xz, cv_w, cv_b, xic, DIN, 2 * DIN, 1);
    // 5) xd[b,k,:,:] = xproj_w[k] @ xs[b,k]  (gathered, never materialized)
    xproj_gemm_wmma<<<dim3(Ll / 16, 3, Bc * K4), 32, 0, stream>>>(xic, xproj_w, xd);
    // 6) selective scan (dt inline)
    scan_kernel<<<(Bc * K4 * DIN) / 16, 256, 0, stream>>>(xic, xd, dtp_w, dtp_b, A_logs, Ds, oy);
    // 7) combine 4 directions -> (B,L,DIN)
    combine_kernel<<<(Bc * Ll * DIN + 255) / 256, 256, 0, stream>>>(oy, yc);
    // 8) layernorm (out-norm), in place
    layernorm_inplace<<<rows / 8, 256, 0, stream>>>(yc, onorm_g, onorm_b, DIN, rows);
    // 9) y * silu(z)
    ymul_kernel<<<(Bc * Ll * DIN + 255) / 256, 256, 0, stream>>>(yc, xz, ymul);
    // 10) @ outp_w
    gemm_wmma<0><<<dim3(INN / 16, rows / 64), 128, 0, stream>>>(ymul, outp_w, nullptr, tbuf, rows, INN, DIN);
    // 11) layernorm, in place
    layernorm_inplace<<<rows / 8, 256, 0, stream>>>(tbuf, norm_g, norm_b, INN, rows);
    // 12) x2 = silu(x @ w2 + b2)
    gemm_wmma<1><<<dim3(INN / 16, rows / 64), 128, 0, stream>>>(x, w2, b2, x2, rows, INN, C0);
    // 13) fin = t * x2
    emul_kernel<<<(rows * INN + 255) / 256, 256, 0, stream>>>(tbuf, x2, fin, rows * INN);
    // 14) out = fin @ wo + bo
    gemm_wmma<0><<<dim3(C0 / 16, rows / 64), 128, 0, stream>>>(fin, wo, bo, out, rows, C0, INN);
}
